// ScRRAMBLeLayer_60301340836146
// MI455X (gfx1250) — matrix operand order repaired and checked by reading, not compiled, not run on hardware
//
#include <hip/hip_runtime.h>

// ---------------------------------------------------------------------------
// ScRRAMBLe layer for MI455X (gfx1250, wave32, WMMA).
//
// Stage 1/3 (routing GEMMs, M=4096 N=16 K={1024,4096}) -> v_wmma_f32_16x16x32_bf16
//   (connectivity is binary -> exact in bf16; f32 accumulate; N=16 matches the
//    WMMA tile exactly, one wave per 16x16 output tile).
// Stage 2/4 (per-core 256x256 block GEMVs, weights unshared -> no WMMA mapping)
//   -> f32 FMA with 4 independent accumulators, LDS-staged vector.
//
// Roofline: ~208 MB streamed / call, ~0.8 GFLOP -> ~4 FLOP/byte, bandwidth
// bound (23.3 TB/s HBM floor ~9 us).
//
// Codegen note: all gathers use ONE per-lane base pointer per matrix with
// compile-time byte offsets (signed 24-bit IOFFSET covers the 64-K window:
// A max 55*16 KB ~ 0.9 MB, B max 55*64 B), advanced by a single 64-bit add
// per iteration -> loads clause cleanly, no per-load VALU address math.
// ---------------------------------------------------------------------------

typedef __attribute__((ext_vector_type(16))) __bf16 v16bf;
typedef __attribute__((ext_vector_type(8)))  float  v8f;

#define SLOT_LEN   16
#define SLOTS      16
#define CORE_LEN   256      // SLOTS * SLOT_LEN
#define N_CORES    256
#define ROW_ELEMS  4096     // (k,l) row length of Ci / C_cores
#define TILE_F     (N_CORES * CORE_LEN)   // 65536 floats per partial buffer

// Deterministic hash-based ~N(0,1) noise (sum of 4 uniforms, CLT), scaled by 0.05.
__device__ __forceinline__ float noisy_term(unsigned idx, unsigned seed) {
  unsigned x = idx + seed * 0x9E3779B9u;
  x ^= x >> 16; x *= 0x7FEB352Du;
  x ^= x >> 15; x *= 0x846CA68Bu;
  x ^= x >> 16;
  float u0 = (float)( x        & 0xFFu);
  float u1 = (float)((x >>  8) & 0xFFu);
  float u2 = (float)((x >> 16) & 0xFFu);
  float u3 = (float)((x >> 24) & 0xFFu);
  float z  = ((u0 + u1 + u2 + u3) * (1.0f / 255.0f) - 2.0f) * 1.7320508f;
  return 0.05f * z;
}

// ---------------------------------------------------------------------------
// Routing stage: out_part[s][k] = sum_{K in split s} C[(ij),(k,l)]^T * B[(ij),m]
//   C : (Ktot, 4096) row-major   (Ci or C_cores, binary 0/1 floats)
//   B : (Ktot, 16)   row-major   (x reshaped, or y1)
//   outp : nsplit partial tensors, each (256, 16, 16) f32
// One wave (32 lanes, EXEC all-ones as WMMA requires) per (k-tile, split).
// Each iteration: 64 constant-offset loads + 2 WMMAs in flight per wave.
// ---------------------------------------------------------------------------
__global__ __launch_bounds__(32) void routing_wmma(
    const float* __restrict__ C,
    const float* __restrict__ B,
    float* __restrict__ outp,
    int Ktot, int nsplit) {
  const int k    = blockIdx.x;       // output core tile (0..255)
  const int s    = blockIdx.y;       // K split
  const int Klen = Ktot / nsplit;    // multiple of 64 for all splits used
  const int K0   = s * Klen;
  const int lane = threadIdx.x;      // 0..31
  const int M    = lane & 15;        // A-row (= l) / D-col (= m) per layouts
  const int half = lane >> 4;

  // Per-lane base pointers with the lane-dependent part (M, half, K0) folded
  // in once; element offsets below are compile-time constants.
  //   A lane layout (16-bit A 16x32): K = K0 + half*8 + {e, e+16} (+32,+48 for
  //   the second chunk).  B lane layout (32x16): K = K0 + half*8 + {e, e+16}.
  const float* Ap = C + (size_t)k * SLOT_LEN + M
                      + (size_t)(K0 + half * 8)  * ROW_ELEMS;
  const float* Bp = B + M
                      + (size_t)(K0 + half * 8)  * SLOT_LEN;

  v8f acc = {};
  for (int it = 0; it < Klen / 64; ++it) {
    v16bf a0, b0, a1, b1;
#pragma unroll
    for (int e = 0; e < 8; ++e) {
      a0[e]     = (__bf16)Ap[(e     ) * ROW_ELEMS];
      a0[e + 8] = (__bf16)Ap[(e + 16) * ROW_ELEMS];
      a1[e]     = (__bf16)Ap[(e + 32) * ROW_ELEMS];
      a1[e + 8] = (__bf16)Ap[(e + 48) * ROW_ELEMS];
      b0[e]     = (__bf16)Bp[(e     ) * SLOT_LEN];
      b0[e + 8] = (__bf16)Bp[(e + 16) * SLOT_LEN];
      b1[e]     = (__bf16)Bp[(e + 32) * SLOT_LEN];
      b1[e + 8] = (__bf16)Bp[(e + 48) * SLOT_LEN];
    }
    acc = __builtin_amdgcn_wmma_f32_16x16x32_bf16(
        false, a0, false, b0, (short)0, acc, false, false);
    acc = __builtin_amdgcn_wmma_f32_16x16x32_bf16(
        false, a1, false, b1, (short)0, acc, false, false);
    Ap += 64 * ROW_ELEMS;     // one 64-bit add per pointer per iteration
    Bp += 64 * SLOT_LEN;
  }

  // D 16x16 f32 layout: lane -> col N = lane&15; VGPR r -> row = half*8 + r.
  float* o = outp + (size_t)s * TILE_F + (size_t)k * CORE_LEN;
#pragma unroll
  for (int r = 0; r < 8; ++r)
    o[(half * 8 + r) * SLOT_LEN + M] = acc[r];
}

// ---------------------------------------------------------------------------
// Per-core block transform + noisy threshold:
//   y[i, j, l] = act( sum_{k,m} W[i,j,k,l,m] * h[i,k,m] )
// One 256-thread block per core; deterministic reduction of K-split partials;
// 4 independent accumulators break the FMA latency chain; constant-offset
// float4 loads off a single base pointer.
// ---------------------------------------------------------------------------
__global__ __launch_bounds__(256) void core_transform(
    const float* __restrict__ W,    // (256,16,16,16,16) f32
    const float* __restrict__ hp,   // nsplit partials, each (256,256)
    float* __restrict__ y,          // (256,256)
    unsigned seed, int nsplit) {
  __shared__ float hs[CORE_LEN];
  const int core = blockIdx.x;
  const int t    = threadIdx.x;     // j = t>>4, l = t&15

  float hv = 0.f;
  for (int s = 0; s < nsplit; ++s)
    hv += hp[(size_t)s * TILE_F + core * CORE_LEN + t];
  hs[t] = hv;
  __syncthreads();

  const int j = t >> 4, l = t & 15;
  const float4* Wr =
      (const float4*)(W + (size_t)core * 65536 + j * 4096 + l * 16);

  float a0 = 0.f, a1 = 0.f, a2 = 0.f, a3 = 0.f;
#pragma unroll 8
  for (int k = 0; k < 16; ++k) {
#pragma unroll
    for (int q = 0; q < 4; ++q) {
      float4 w = Wr[k * 64 + q];            // const byte offset: k*1024 + q*16
      const float* hk = hs + k * 16 + q * 4;
      a0 = fmaf(w.x, hk[0], a0);
      a1 = fmaf(w.y, hk[1], a1);
      a2 = fmaf(w.z, hk[2], a2);
      a3 = fmaf(w.w, hk[3], a3);
    }
  }
  const float acc = (a0 + a1) + (a2 + a3);

  const unsigned idx = (unsigned)(core * CORE_LEN + t);
  const float nz = noisy_term(idx, seed);
  y[(size_t)core * CORE_LEN + t] = (acc + nz > 0.f) ? acc : 0.f;
}

// ---------------------------------------------------------------------------
// Launch: inputs in setup_inputs() order: x, Wi, Wo, Ci, C_cores (all f32).
// d_ws layout: [nsplit x 65536] hpart | [65536] y1 | [nsplit x 65536] h2part.
// nsplit chosen from ws_size (fixed across calls -> deterministic):
//   8 -> 4.25 MB, halving down to 1 (K stays 64-aligned for every choice).
// ---------------------------------------------------------------------------
extern "C" void kernel_launch(void* const* d_in, const int* in_sizes, int n_in,
                              void* d_out, int out_size, void* d_ws, size_t ws_size,
                              hipStream_t stream) {
  const float* x  = (const float*)d_in[0];
  const float* Wi = (const float*)d_in[1];
  const float* Wo = (const float*)d_in[2];
  const float* Ci = (const float*)d_in[3];
  const float* Cc = (const float*)d_in[4];
  float* out = (float*)d_out;

  int nsplit = 8;
  while (nsplit > 1 &&
         (size_t)(2 * nsplit + 1) * TILE_F * sizeof(float) > ws_size)
    nsplit >>= 1;

  float* ws     = (float*)d_ws;
  float* hpart  = ws;
  float* y1     = ws + (size_t)nsplit * TILE_F;
  float* h2part = y1 + TILE_F;

  dim3 gridR(N_CORES, nsplit);
  routing_wmma  <<<gridR,   32, 0, stream>>>(Ci, x,  hpart, 1024, nsplit);
  core_transform<<<N_CORES, 256, 0, stream>>>(Wi, hpart, y1, 0x13579BDFu, nsplit);
  routing_wmma  <<<gridR,   32, 0, stream>>>(Cc, y1, h2part, 4096, nsplit);
  core_transform<<<N_CORES, 256, 0, stream>>>(Wo, h2part, out, 0x2468ACE1u, nsplit);
}
